// MultiHeadAttention_57887569216003
// MI455X (gfx1250) — matrix operand order repaired and checked by reading, not compiled, not run on hardware
//
#include <hip/hip_runtime.h>
#include <hip/hip_bf16.h>

// ---------------------------------------------------------------------------
// MHA forward for MI455X (gfx1250, wave32, WMMA bf16 -> f32 accum)
//   B=4, S=2048, D_MODEL=1024, H=16, d_k=64
// Pipeline: cvt f32->bf16  ->  Q/K/V projection GEMMs (WMMA; Q pre-scaled by
//           1/sqrt(dk)*log2(e))  ->  flash attention (async LDS staging of
//           K/V blocks via saddr-form global_load_async_to_lds, WMMA QK^T +
//           base-2 online softmax with DPP row_xmask reductions + WMMA P*V)
//           ->  output projection GEMM (WMMA, f32 out)
// ---------------------------------------------------------------------------

typedef __attribute__((ext_vector_type(16))) __bf16 v16bf;
typedef __attribute__((ext_vector_type(8)))  float  v8f;

union Frag16 { v16bf v; uint4 q[2]; };

__device__ __forceinline__ unsigned short f2bf_rne(float x) {
    unsigned u = __float_as_uint(x);
    u += 0x7fffu + ((u >> 16) & 1u);           // round-to-nearest-even
    return (unsigned short)(u >> 16);
}

// XOR-lane butterfly within each 16-lane row via DPP16 ROW_XMASK (VALU, no LDS)
template <int MASK>
__device__ __forceinline__ float dppxor(float x) {
    return __int_as_float(__builtin_amdgcn_update_dpp(
        0, __float_as_int(x), 0x160 | MASK, 0xF, 0xF, true));
}
__device__ __forceinline__ float redmax16(float x) {
    x = fmaxf(x, dppxor<1>(x));
    x = fmaxf(x, dppxor<2>(x));
    x = fmaxf(x, dppxor<4>(x));
    x = fmaxf(x, dppxor<8>(x));
    return x;                                   // reduced within each 16-lane half
}
__device__ __forceinline__ float redsum16(float x) {
    x += dppxor<1>(x);
    x += dppxor<2>(x);
    x += dppxor<4>(x);
    x += dppxor<8>(x);
    return x;
}

// ---------------------------------------------------------------------------
// f32 -> bf16 (vectorized, RNE)
// ---------------------------------------------------------------------------
__global__ void cvt_f32_bf16_kernel(const float4* __restrict__ src,
                                    ushort4* __restrict__ dst, int n4) {
    int i = blockIdx.x * blockDim.x + threadIdx.x;
    if (i < n4) {
        float4 f = src[i];
        ushort4 o;
        o.x = f2bf_rne(f.x); o.y = f2bf_rne(f.y);
        o.z = f2bf_rne(f.z); o.w = f2bf_rne(f.w);
        dst[i] = o;
    }
}

// ---------------------------------------------------------------------------
// GEMM:  C[M,N] = A[M,K] * W[N,K]^T + bias   (M=8192, N=K=1024, bf16 in, f32 acc)
// MODE 0: store bf16 row-major [M,N]
// MODE 1: store bf16 transposed-per-head:  Vt[(b*16+h)*64 + d][s]   (for P*V)
// MODE 2: store f32 row-major (final output)
// MODE 3: store bf16 row-major, scaled by 1/sqrt(dk)*log2(e)  (Q projection)
// Block = 128 thr (4 waves, 2x2), block tile 128x128, wave tile 64x64.
// ---------------------------------------------------------------------------
template <int MODE>
__global__ __launch_bounds__(128)
void gemm_bf16_wmma(const unsigned short* __restrict__ A,
                    const unsigned short* __restrict__ W,
                    const float* __restrict__ bias,
                    void* __restrict__ Out) {
    constexpr int K = 1024, N = 1024;
    const int lane = threadIdx.x & 31;
    const int wid  = threadIdx.x >> 5;
    const int lr   = lane & 15;      // row/col within fragment
    const int lg   = lane >> 4;      // half-wave group
    const int row0 = blockIdx.y * 128 + (wid >> 1) * 64;
    const int col0 = blockIdx.x * 128 + (wid & 1) * 64;

    v8f z = {};
    v8f c[4][4];
#pragma unroll
    for (int i = 0; i < 4; ++i)
#pragma unroll
        for (int j = 0; j < 4; ++j) c[i][j] = z;

    for (int kb = 0; kb < K; kb += 32) {
        Frag16 a[4], b[4];
#pragma unroll
        for (int mi = 0; mi < 4; ++mi) {   // A: lane holds row, split K chunks
            const unsigned short* p = A + (size_t)(row0 + mi * 16 + lr) * K + kb;
            a[mi].q[0] = *(const uint4*)(p + lg * 8);
            a[mi].q[1] = *(const uint4*)(p + 16 + lg * 8);
        }
#pragma unroll
        for (int ni = 0; ni < 4; ++ni) {   // B: lane holds col, split K chunks
            const unsigned short* p = W + (size_t)(col0 + ni * 16 + lr) * K + kb;
            b[ni].q[0] = *(const uint4*)(p + lg * 8);
            b[ni].q[1] = *(const uint4*)(p + 16 + lg * 8);
        }
#pragma unroll
        for (int mi = 0; mi < 4; ++mi)
#pragma unroll
            for (int ni = 0; ni < 4; ++ni)
                c[mi][ni] = __builtin_amdgcn_wmma_f32_16x16x32_bf16(
                    false, a[mi].v, false, b[ni].v, (short)0, c[mi][ni], false, false);
    }

#pragma unroll
    for (int mi = 0; mi < 4; ++mi) {
#pragma unroll
        for (int ni = 0; ni < 4; ++ni) {
            const int n  = col0 + ni * 16 + lr;
            const float bv = bias[n];
#pragma unroll
            for (int r = 0; r < 8; ++r) {
                const int m = row0 + mi * 16 + lg * 8 + r;
                const float val = c[mi][ni][r] + bv;
                if (MODE == 0) {
                    ((unsigned short*)Out)[(size_t)m * N + n] = f2bf_rne(val);
                } else if (MODE == 1) {
                    const int bb = m >> 11, s = m & 2047;   // m = b*2048 + s
                    const int h  = n >> 6,  d = n & 63;     // n = h*64 + d
                    ((unsigned short*)Out)[(size_t)((bb * 16 + h) * 64 + d) * 2048 + s] =
                        f2bf_rne(val);
                } else if (MODE == 3) {
                    // fold softmax scale (1/8) and log2(e) into Q
                    ((unsigned short*)Out)[(size_t)m * N + n] =
                        f2bf_rne(val * 0.18033688011112042f);
                } else {
                    ((float*)Out)[(size_t)m * N + n] = val;
                }
            }
        }
    }
}

// ---------------------------------------------------------------------------
// Flash attention: per block = one (b,h) x 64 query rows; 4 waves x 16 rows.
// K/V 64x64 blocks staged to LDS with saddr-form async copies (double-
// buffered, ASYNCcnt); base-2 online softmax; P re-laid out via LDS.
// ---------------------------------------------------------------------------
__global__ __launch_bounds__(128)
void flash_attn_wmma(const unsigned short* __restrict__ Qb,  // [B*S,1024] bf16 (pre-scaled)
                     const unsigned short* __restrict__ Kb,  // [B*S,1024] bf16
                     const unsigned short* __restrict__ Vt,  // [B*H*64,2048] bf16
                     unsigned short* __restrict__ Ctx) {     // [B*S,1024] bf16
    constexpr int S = 2048, D = 1024, DK = 64, NJ = S / 64;
    __shared__ __align__(16) unsigned short Kl[2][64 * 64];  // 16 KB
    __shared__ __align__(16) unsigned short Vl[2][64 * 64];  // 16 KB
    __shared__ __align__(16) unsigned short Pl[4][16 * 64];  // 8 KB, wave-private

    const int tid  = threadIdx.x;
    const int lane = tid & 31;
    const int wid  = tid >> 5;
    const int lr   = lane & 15, lg = lane >> 4;
    const int bh   = blockIdx.y;                 // 0..63
    const int b    = bh >> 4, h = bh & 15;
    const int q0   = blockIdx.x * 64 + wid * 16; // query row in [0,S)

    const unsigned short* Kbase = Kb + (size_t)b * S * D + h * DK;
    const unsigned short* Vbase = Vt + (size_t)bh * 64 * S;  // row = d, contiguous in s

    // loop-invariant per-thread copy offsets (512 16B chunks per 8KB tile)
    unsigned kvo[4], vvo[4], ldsK[4], ldsV[4];
#pragma unroll
    for (int r = 0; r < 4; ++r) {
        const int c   = r * 128 + tid;
        const int row = c >> 3;
        const int off = (c & 7) * 8;
        kvo[r]  = (unsigned)((row * D + off) * 2);   // bytes into K slab (row stride D)
        vvo[r]  = (unsigned)((row * S + off) * 2);   // bytes into V slab (row stride S)
        ldsK[r] = (unsigned)(size_t)&Kl[0][row * 64 + off];
        ldsV[r] = (unsigned)(size_t)&Vl[0][row * 64 + off];
    }
    const unsigned long long kSlab = (unsigned long long)(size_t)Kbase;
    const unsigned long long vSlab = (unsigned long long)(size_t)Vbase;

    // cooperative async stage of one 64-key block of K and V into buffer `buf`
    auto prefetch = [&](int buf, int jb) {
        const unsigned long long kb = kSlab + (unsigned long long)jb * (64ull * D * 2ull);
        const unsigned long long vb = vSlab + (unsigned long long)jb * (64ull * 2ull);
        const unsigned bo = (unsigned)(buf << 13);
#pragma unroll
        for (int r = 0; r < 4; ++r) {
            asm volatile("global_load_async_to_lds_b128 %0, %1, %2"
                         :: "v"(ldsK[r] + bo), "v"(kvo[r]), "s"(kb) : "memory");
            asm volatile("global_load_async_to_lds_b128 %0, %1, %2"
                         :: "v"(ldsV[r] + bo), "v"(vvo[r]), "s"(vb) : "memory");
        }
    };

    // Q A-fragments (16 rows x 64 = 2 frags), resident whole kernel
    Frag16 qf[2];
    {
        const unsigned short* p = Qb + (size_t)(b * S + q0 + lr) * D + h * DK;
#pragma unroll
        for (int kf = 0; kf < 2; ++kf) {
            qf[kf].q[0] = *(const uint4*)(p + kf * 32 + lg * 8);
            qf[kf].q[1] = *(const uint4*)(p + kf * 32 + 16 + lg * 8);
        }
    }

    v8f z = {};
    v8f o[4];
#pragma unroll
    for (int t = 0; t < 4; ++t) o[t] = z;
    float mrow[8], lrow[8];
#pragma unroll
    for (int r = 0; r < 8; ++r) { mrow[r] = -1e30f; lrow[r] = 0.f; }

    prefetch(0, 0);

    for (int jb = 0; jb < NJ; ++jb) {
        const int cur = jb & 1;
        if (jb + 1 < NJ) {
            prefetch(cur ^ 1, jb + 1);                       // overlap next block copy
            asm volatile("s_wait_asynccnt 0x8" ::: "memory"); // current block done
        } else {
            asm volatile("s_wait_asynccnt 0x0" ::: "memory");
        }
        __syncthreads();   // all waves' async chunks of current buffer visible

        // ---- S = Q * K^T (scale folded into Q) : 4 frags over 64 keys ----
        v8f s[4];
#pragma unroll
        for (int t = 0; t < 4; ++t) {
            s[t] = z;
#pragma unroll
            for (int kf = 0; kf < 2; ++kf) {
                Frag16 bf;
                const unsigned short* p = &Kl[cur][(t * 16 + lr) * 64 + kf * 32];
                bf.q[0] = *(const uint4*)(p + lg * 8);
                bf.q[1] = *(const uint4*)(p + 16 + lg * 8);
                s[t] = __builtin_amdgcn_wmma_f32_16x16x32_bf16(
                    false, qf[kf].v, false, bf.v, (short)0, s[t], false, false);
            }
        }

        // ---- base-2 online softmax (row stats per-vgpr, per 16-lane half) ----
        float nm[8], corr[8];
#pragma unroll
        for (int r = 0; r < 8; ++r) {
            float mx = fmaxf(fmaxf(s[0][r], s[1][r]), fmaxf(s[2][r], s[3][r]));
            mx = redmax16(mx);
            nm[r]   = fmaxf(mrow[r], mx);
            corr[r] = exp2f(mrow[r] - nm[r]);
        }
#pragma unroll
        for (int t = 0; t < 4; ++t)
#pragma unroll
            for (int r = 0; r < 8; ++r) s[t][r] = exp2f(s[t][r] - nm[r]);
#pragma unroll
        for (int r = 0; r < 8; ++r) {
            float rs = redsum16(s[0][r] + s[1][r] + s[2][r] + s[3][r]);
            lrow[r] = lrow[r] * corr[r] + rs;
            mrow[r] = nm[r];
        }
#pragma unroll
        for (int t = 0; t < 4; ++t)
#pragma unroll
            for (int r = 0; r < 8; ++r) o[t][r] *= corr[r];

        // ---- P: C-layout -> LDS -> A-layout fragments (wave-private tile) ----
#pragma unroll
        for (int t = 0; t < 4; ++t)
#pragma unroll
            for (int r = 0; r < 8; ++r)
                Pl[wid][(lg * 8 + r) * 64 + t * 16 + lr] = f2bf_rne(s[t][r]);
        asm volatile("s_wait_dscnt 0x0" ::: "memory");

        Frag16 pf[2];
#pragma unroll
        for (int kf = 0; kf < 2; ++kf) {
            pf[kf].q[0] = *(const uint4*)&Pl[wid][lr * 64 + kf * 32 + lg * 8];
            pf[kf].q[1] = *(const uint4*)&Pl[wid][lr * 64 + kf * 32 + 16 + lg * 8];
        }

        // ---- O += P * V  (V rows = d, contiguous in keys, from LDS) ----
#pragma unroll
        for (int t = 0; t < 4; ++t) {
#pragma unroll
            for (int kf = 0; kf < 2; ++kf) {
                Frag16 vf;
                const unsigned short* p = &Vl[cur][(t * 16 + lr) * 64 + kf * 32];
                vf.q[0] = *(const uint4*)(p + lg * 8);
                vf.q[1] = *(const uint4*)(p + 16 + lg * 8);
                o[t] = __builtin_amdgcn_wmma_f32_16x16x32_bf16(
                    false, pf[kf].v, false, vf.v, (short)0, o[t], false, false);
            }
        }

        __syncthreads();   // all waves done reading `cur` before it is re-staged
    }

    // ---- normalize and store ctx (bf16) ----
#pragma unroll
    for (int r = 0; r < 8; ++r) lrow[r] = 1.f / lrow[r];
#pragma unroll
    for (int t = 0; t < 4; ++t)
#pragma unroll
        for (int r = 0; r < 8; ++r) {
            const int m = q0 + lg * 8 + r;
            const int n = h * DK + t * 16 + lr;
            Ctx[(size_t)(b * S + m) * D + n] = f2bf_rne(o[t][r] * lrow[r]);
        }
}

// ---------------------------------------------------------------------------
extern "C" void kernel_launch(void* const* d_in, const int* in_sizes, int n_in,
                              void* d_out, int out_size, void* d_ws, size_t ws_size,
                              hipStream_t stream) {
    const float* q  = (const float*)d_in[0];
    const float* k  = (const float*)d_in[1];
    const float* v  = (const float*)d_in[2];
    const float* wq = (const float*)d_in[3];
    const float* bq = (const float*)d_in[4];
    const float* wk = (const float*)d_in[5];
    const float* bk = (const float*)d_in[6];
    const float* wv = (const float*)d_in[7];
    const float* bv = (const float*)d_in[8];
    const float* wo = (const float*)d_in[9];
    const float* bo = (const float*)d_in[10];

    const size_t MS  = 8192ull * 1024ull;   // tokens * d_model
    const size_t WSZ = 1024ull * 1024ull;   // weight elements

    unsigned short* base = (unsigned short*)d_ws;
    unsigned short* Xq  = base;        // bf16 inputs
    unsigned short* Xk  = Xq  + MS;
    unsigned short* Xv  = Xk  + MS;
    unsigned short* Wqb = Xv  + MS;    // bf16 weights
    unsigned short* Wkb = Wqb + WSZ;
    unsigned short* Wvb = Wkb + WSZ;
    unsigned short* Wob = Wvb + WSZ;
    unsigned short* Qb  = Wob + WSZ;   // projected Q/K (bf16)
    unsigned short* Kb  = Qb  + MS;
    unsigned short* Vt  = Kb  + MS;    // V transposed per head: [B*H*64, 2048]
    unsigned short* Ctx = Xq;          // reuse: Xq is dead after Q projection

    auto cvt = [&](const float* s, unsigned short* d, size_t n) {
        int n4 = (int)(n / 4);
        cvt_f32_bf16_kernel<<<(n4 + 255) / 256, 256, 0, stream>>>(
            (const float4*)s, (ushort4*)d, n4);
    };
    cvt(q,  Xq,  MS);  cvt(k,  Xk,  MS);  cvt(v,  Xv,  MS);
    cvt(wq, Wqb, WSZ); cvt(wk, Wkb, WSZ); cvt(wv, Wvb, WSZ); cvt(wo, Wob, WSZ);

    dim3 ggrid(8, 64), gblk(128);
    gemm_bf16_wmma<3><<<ggrid, gblk, 0, stream>>>(Xq, Wqb, bq, Qb);  // Q (pre-scaled)
    gemm_bf16_wmma<0><<<ggrid, gblk, 0, stream>>>(Xk, Wkb, bk, Kb);  // K
    gemm_bf16_wmma<1><<<ggrid, gblk, 0, stream>>>(Xv, Wvb, bv, Vt);  // V (transposed)

    flash_attn_wmma<<<dim3(32, 64), 128, 0, stream>>>(Qb, Kb, Vt, Ctx);

    gemm_bf16_wmma<2><<<ggrid, gblk, 0, stream>>>(Ctx, Wob, bo, d_out);
}
